// RowDecoder_8718783611086
// MI455X (gfx1250) — matrix operand order repaired and checked by reading, not compiled, not run on hardware
//
#include <hip/hip_runtime.h>
#include <hip/hip_bf16.h>

// ---------------- CDNA5 WMMA types ----------------
typedef __attribute__((ext_vector_type(16))) __bf16 v16bf;
typedef __attribute__((ext_vector_type(8)))  float  v8f;

// Problem constants
#define Bsz   256
#define Hdim  512
#define Edim  512
#define Vdim  32000
#define FEATd 512
#define HWd   256   // 16*16 spatial

// ------------------------------------------------------------------
// Fragment loaders (CDNA5 ISA 7.12.2 wave32 layouts, bf16 WMMA 16x16x32):
//   A frag (16x32): lane -> M = lane&15 ; half = lane>>4
//     frag[0..7]  = K(k0 + half*8  + j)
//     frag[8..15] = K(k0 + 16 + half*8 + j)
//   B frag (32x16): lane -> N = lane&15
//     frag[0..7]  = K(k0 + half*8  + j), frag[8..15] = K(k0 + 16 + half*8 + j)
//   C/D f32: vgpr v -> M = v + half*8, N = lane&15
// ------------------------------------------------------------------

// A frag, k-contiguous row: two 8-float runs at +0 and +16 -> 4x float4 loads
__device__ __forceinline__ v16bf load_frag_a_contig(const float* __restrict__ p)
{
  const float4* q = (const float4*)p;          // p = Arow + ka (16B aligned)
  const float4 x0 = q[0], x1 = q[1], x2 = q[4], x3 = q[5];
  v16bf f;
  f[0]=(__bf16)x0.x; f[1]=(__bf16)x0.y; f[2]=(__bf16)x0.z; f[3]=(__bf16)x0.w;
  f[4]=(__bf16)x1.x; f[5]=(__bf16)x1.y; f[6]=(__bf16)x1.z; f[7]=(__bf16)x1.w;
  f[8]=(__bf16)x2.x; f[9]=(__bf16)x2.y; f[10]=(__bf16)x2.z; f[11]=(__bf16)x2.w;
  f[12]=(__bf16)x3.x; f[13]=(__bf16)x3.y; f[14]=(__bf16)x3.z; f[15]=(__bf16)x3.w;
  return f;
}

// A frag with strided k (attention: A[i + f*HW])
__device__ __forceinline__ v16bf load_frag_a_strided(const float* __restrict__ Arow,
                                                     int ka, long s)
{
  v16bf f;
#pragma unroll
  for (int j = 0; j < 8; ++j) {
    f[j]     = (__bf16)Arow[(long)(ka + j)      * s];
    f[j + 8] = (__bf16)Arow[(long)(ka + 16 + j) * s];
  }
  return f;
}

// B frag: two 8-float runs at +0 and +16 -> 4x float4 loads
__device__ __forceinline__ v16bf load_frag_b(const float* __restrict__ p)
{
  const float4* q = (const float4*)p;          // p = Wrow + kb (16B aligned)
  const float4 x0 = q[0], x1 = q[1], x2 = q[4], x3 = q[5];
  v16bf f;
  f[0]=(__bf16)x0.x; f[1]=(__bf16)x0.y; f[2]=(__bf16)x0.z; f[3]=(__bf16)x0.w;
  f[4]=(__bf16)x1.x; f[5]=(__bf16)x1.y; f[6]=(__bf16)x1.z; f[7]=(__bf16)x1.w;
  f[8]=(__bf16)x2.x; f[9]=(__bf16)x2.y; f[10]=(__bf16)x2.z; f[11]=(__bf16)x2.w;
  f[12]=(__bf16)x3.x; f[13]=(__bf16)x3.y; f[14]=(__bf16)x3.z; f[15]=(__bf16)x3.w;
  return f;
}

#define WMMA_BF16(a, b, c) \
  __builtin_amdgcn_wmma_f32_16x16x32_bf16(false, (a), false, (b), (short)0, (c), false, false)

// ------------------------------------------------------------------
// 32x64 macro-tile per wave: acc[2][4], 8 WMMAs per k-step.
// A-frags amortized 4x, B-frags 2x  (vs 1x/1x for naive 16x16).
// ------------------------------------------------------------------
__device__ __forceinline__ void gemm_macro_tile(
    const float* __restrict__ A, const float* __restrict__ W,
    int K, long ldw, int rowBase, int colBase, v8f acc[2][4])
{
  const int lane = threadIdx.x & 31;
  const int half = lane >> 4;
  const int l16  = lane & 15;
  const float* Ar0 = A + (long)(rowBase + l16)      * K;
  const float* Ar1 = A + (long)(rowBase + 16 + l16) * K;
  const float* Wr  = W + (long)(colBase + l16)      * ldw;
  for (int k0 = 0; k0 < K; k0 += 32) {
    const int ka = k0 + half * 8;
    const int kb = k0 + half * 8;
    const v16bf a0 = load_frag_a_contig(Ar0 + ka);
    const v16bf a1 = load_frag_a_contig(Ar1 + ka);
#pragma unroll
    for (int c = 0; c < 4; ++c) {
      const v16bf bf = load_frag_b(Wr + (long)c * 16 * ldw + kb);
      acc[0][c] = WMMA_BF16(a0, bf, acc[0][c]);
      acc[1][c] = WMMA_BF16(a1, bf, acc[1][c]);
    }
  }
}

// ---------------- generic  C = A @ W^T + bias ----------------
// grid = (M/32, N/64), block = 32 (one wave per 32x64 macro-tile)
__global__ void gemm_bias_kernel(const float* __restrict__ A,
                                 const float* __restrict__ W,
                                 const float* __restrict__ bias,
                                 float* __restrict__ C, int N, int K)
{
  const int tM = blockIdx.x, tN = blockIdx.y;
  v8f acc[2][4] = {};
  gemm_macro_tile(A, W, K, K, tM * 32, tN * 64, acc);
  const int lane = threadIdx.x & 31, half = lane >> 4, l16 = lane & 15;
#pragma unroll
  for (int c = 0; c < 4; ++c) {
    const int col = tN * 64 + c * 16 + l16;
    const float bb = bias ? bias[col] : 0.0f;
#pragma unroll
    for (int r = 0; r < 2; ++r) {
#pragma unroll
      for (int v = 0; v < 8; ++v) {
        const int row = tM * 32 + r * 16 + half * 8 + v;
        C[(long)row * N + col] = acc[r][c][v] + bb;
      }
    }
  }
}

// ---------------- attention scores (fused proj + tanh + partial H-reduce) ---
// proj[b,i,h] = sum_f flat[b,f,i] * W_F_w[h,f]
// part[(b*HW+i)*8 + tN] = sum_{h in 64-col block} tanh(h_t[b,h]+W_F_b[h]+proj)
// Deterministic (no FP atomics). grid = (HW/16, H/64, B), block = 32.
__global__ void attn_scores_kernel(const float* __restrict__ flat,
                                   const float* __restrict__ h_t,
                                   const float* __restrict__ WFw,
                                   const float* __restrict__ WFb,
                                   float* __restrict__ part)
{
  const int tI = blockIdx.x, tN = blockIdx.y, b = blockIdx.z;
  const int lane = threadIdx.x & 31, half = lane >> 4, l16 = lane & 15;
  const float* A    = flat + (long)b * FEATd * HWd;      // element (i,f) = A[i + f*HW]
  const float* Arow = A + tI * 16 + l16;
  const float* Wr   = WFw + (long)(tN * 64 + l16) * FEATd;
  v8f acc[4] = {};
  for (int k0 = 0; k0 < FEATd; k0 += 32) {
    const v16bf af = load_frag_a_strided(Arow, k0 + half * 8, HWd);
    const int kb = k0 + half * 8;
#pragma unroll
    for (int c = 0; c < 4; ++c) {
      const v16bf bf = load_frag_b(Wr + (long)c * 16 * FEATd + kb);
      acc[c] = WMMA_BF16(af, bf, acc[c]);
    }
  }
  // add h_t + bias, tanh, reduce over the 64 h-columns of this block
  float addc[4];
#pragma unroll
  for (int c = 0; c < 4; ++c) {
    const int hcol = tN * 64 + c * 16 + l16;
    addc[c] = h_t[(long)b * Hdim + hcol] + WFb[hcol];
  }
#pragma unroll
  for (int v = 0; v < 8; ++v) {
    float t = 0.0f;
#pragma unroll
    for (int c = 0; c < 4; ++c) t += tanhf(acc[c][v] + addc[c]);
    t += __shfl_xor(t, 1, 32);            // reduce the 16 lanes of each half
    t += __shfl_xor(t, 2, 32);
    t += __shfl_xor(t, 4, 32);
    t += __shfl_xor(t, 8, 32);
    if (l16 == 0) {
      const int row = tI * 16 + half * 8 + v;
      part[((long)b * HWd + row) * 8 + tN] = t;
    }
  }
}

// ---------------- softmax over HW (sums the 8 H-block partials) -------------
__global__ void attn_softmax_kernel(const float* __restrict__ part,
                                    float* __restrict__ attn)
{
  const int b = blockIdx.x, i = threadIdx.x;      // 256 threads
  __shared__ float red[HWd];
  const float* pp = part + ((long)b * HWd + i) * 8;
  float s = 0.0f;
#pragma unroll
  for (int j = 0; j < 8; ++j) s += pp[j];
  const float v = s * (1.0f / ((float)Hdim * 0.1f));   // mean over H, then /0.1
  red[i] = v; __syncthreads();
  for (int st = 128; st > 0; st >>= 1) { if (i < st) red[i] = fmaxf(red[i], red[i + st]); __syncthreads(); }
  const float m = red[0]; __syncthreads();
  const float e = expf(v - m);
  red[i] = e; __syncthreads();
  for (int st = 128; st > 0; st >>= 1) { if (i < st) red[i] += red[i + st]; __syncthreads(); }
  attn[(long)b * HWd + i] = e / red[0];
}

// ---------------- context[b,f] = sum_i attn[b,i] * flat[b,f,i] --------------
__global__ void context_kernel(const float* __restrict__ flat,
                               const float* __restrict__ attn,
                               float* __restrict__ ctx)
{
  const int b = blockIdx.x, t = threadIdx.x;      // 256 threads
  __shared__ float a[HWd];
  a[t] = attn[(long)b * HWd + t];
  __syncthreads();
  for (int f = t; f < FEATd; f += 256) {
    const float* row = flat + ((long)b * FEATd + f) * HWd;
    float s = 0.0f;
#pragma unroll 4
    for (int i = 0; i < HWd; ++i) s += row[i] * a[i];
    ctx[(long)b * FEATd + f] = s;
  }
}

// ---------------- fused LSTM gates: tok|ctx through W_ih, hid through W_hh --
// grid = (B/32, 4H/64), block = 32
__global__ void gates_kernel(const float* __restrict__ tok,
                             const float* __restrict__ ctx,
                             const float* __restrict__ hid,
                             const float* __restrict__ Wih,
                             const float* __restrict__ bih,
                             const float* __restrict__ Whh,
                             const float* __restrict__ bhh,
                             float* __restrict__ gates)
{
  const int tM = blockIdx.x, tN = blockIdx.y;
  const int lane = threadIdx.x & 31, half = lane >> 4, l16 = lane & 15;
  v8f acc[2][4] = {};
  // three K-segments; W_ih is [4H, E+H] with tok cols [0,E) and ctx cols [E,E+H)
  {
    const float* Ar0 = tok + (long)(tM * 32 + l16)      * Edim;
    const float* Ar1 = tok + (long)(tM * 32 + 16 + l16) * Edim;
    const float* Wr  = Wih + (long)(tN * 64 + l16) * (Edim + Hdim);
    for (int k0 = 0; k0 < Edim; k0 += 32) {
      const v16bf a0 = load_frag_a_contig(Ar0 + k0 + half * 8);
      const v16bf a1 = load_frag_a_contig(Ar1 + k0 + half * 8);
#pragma unroll
      for (int c = 0; c < 4; ++c) {
        const v16bf bf = load_frag_b(Wr + (long)c * 16 * (Edim + Hdim) + k0 + half * 8);
        acc[0][c] = WMMA_BF16(a0, bf, acc[0][c]);
        acc[1][c] = WMMA_BF16(a1, bf, acc[1][c]);
      }
    }
  }
  {
    const float* Ar0 = ctx + (long)(tM * 32 + l16)      * Hdim;
    const float* Ar1 = ctx + (long)(tM * 32 + 16 + l16) * Hdim;
    const float* Wr  = Wih + Edim + (long)(tN * 64 + l16) * (Edim + Hdim);
    for (int k0 = 0; k0 < Hdim; k0 += 32) {
      const v16bf a0 = load_frag_a_contig(Ar0 + k0 + half * 8);
      const v16bf a1 = load_frag_a_contig(Ar1 + k0 + half * 8);
#pragma unroll
      for (int c = 0; c < 4; ++c) {
        const v16bf bf = load_frag_b(Wr + (long)c * 16 * (Edim + Hdim) + k0 + half * 8);
        acc[0][c] = WMMA_BF16(a0, bf, acc[0][c]);
        acc[1][c] = WMMA_BF16(a1, bf, acc[1][c]);
      }
    }
  }
  gemm_macro_tile(hid, Whh, Hdim, Hdim, tM * 32, tN * 64, acc);
#pragma unroll
  for (int c = 0; c < 4; ++c) {
    const int col = tN * 64 + c * 16 + l16;
    const float bb = bih[col] + bhh[col];
#pragma unroll
    for (int r = 0; r < 2; ++r) {
#pragma unroll
      for (int v = 0; v < 8; ++v) {
        const int row = tM * 32 + r * 16 + half * 8 + v;
        gates[(long)row * (4 * Hdim) + col] = acc[r][c][v] + bb;
      }
    }
  }
}

// ---------------- LSTM elementwise ----------------
__global__ void lstm_kernel(const float* __restrict__ gates,
                            const float* __restrict__ cell,
                            float* __restrict__ hNew,
                            float* __restrict__ cNew)
{
  const int idx = blockIdx.x * blockDim.x + threadIdx.x;
  if (idx >= Bsz * Hdim) return;
  const int b = idx >> 9, h = idx & (Hdim - 1);
  const float* g = gates + (long)b * (4 * Hdim);
  const float ig = g[h], fg = g[Hdim + h], gg = g[2 * Hdim + h], og = g[3 * Hdim + h];
  const float sI = 1.0f / (1.0f + expf(-ig));
  const float sF = 1.0f / (1.0f + expf(-fg));
  const float sO = 1.0f / (1.0f + expf(-og));
  const float c  = sF * cell[idx] + sI * tanhf(gg);
  const float hh = sO * tanhf(c);
  cNew[idx] = c;
  hNew[idx] = hh;
}

// ---------------- softmax over V=32000 ----------------
__global__ void vocab_softmax_kernel(const float* __restrict__ logits,
                                     float* __restrict__ probs)
{
  const int b = blockIdx.x, t = threadIdx.x;      // 1024 threads
  __shared__ float red[1024];
  const float* row = logits + (long)b * Vdim;
  float m = -INFINITY;
  for (int j = t; j < Vdim; j += 1024) m = fmaxf(m, row[j]);
  red[t] = m; __syncthreads();
  for (int s = 512; s > 0; s >>= 1) { if (t < s) red[t] = fmaxf(red[t], red[t + s]); __syncthreads(); }
  m = red[0]; __syncthreads();
  float sum = 0.0f;
  for (int j = t; j < Vdim; j += 1024) sum += expf(row[j] - m);
  red[t] = sum; __syncthreads();
  for (int s = 512; s > 0; s >>= 1) { if (t < s) red[t] += red[t + s]; __syncthreads(); }
  const float inv = 1.0f / red[0];
  float* prow = probs + (long)b * Vdim;
  for (int j = t; j < Vdim; j += 1024) prow[j] = expf(row[j] - m) * inv;
}

// ------------------------------------------------------------------
extern "C" void kernel_launch(void* const* d_in, const int* in_sizes, int n_in,
                              void* d_out, int out_size, void* d_ws, size_t ws_size,
                              hipStream_t stream)
{
  const float* tok  = (const float*)d_in[0];   // [B,E]
  const float* hid  = (const float*)d_in[1];   // [B,H]
  const float* cell = (const float*)d_in[2];   // [B,H]
  const float* img  = (const float*)d_in[3];   // [B,F,GH,GW]
  const float* Whw  = (const float*)d_in[4];   // [H,H]
  const float* Whb  = (const float*)d_in[5];   // [H]
  const float* WFw  = (const float*)d_in[6];   // [H,F]
  const float* WFb  = (const float*)d_in[7];   // [H]
  const float* Wih  = (const float*)d_in[8];   // [4H, E+H]
  const float* bih  = (const float*)d_in[9];   // [4H]
  const float* Whh  = (const float*)d_in[10];  // [4H, H]
  const float* bhh  = (const float*)d_in[11];  // [4H]
  const float* Wout = (const float*)d_in[12];  // [V,H]
  const float* bout = (const float*)d_in[13];  // [V]

  float* out = (float*)d_out;
  float* ws  = (float*)d_ws;

  // workspace layout (floats): ~5.5 MB total
  float* ht    = ws;                              // [B,H]        131072
  float* part  = ws + 131072;                     // [B,HW,8]     524288
  float* attn  = ws + 655360;                     // [B,HW]        65536
  float* ctx   = ws + 720896;                     // [B,H]        131072
  float* gates = ws + 851968;                     // [B,4H]       524288

  // output layout: probs | h_new | c_new | logits
  float* probs  = out;                            // B*V = 8,192,000
  float* hNew   = out + (long)Bsz * Vdim;         // B*H
  float* cNew   = hNew + (long)Bsz * Hdim;        // B*H
  float* logits = cNew + (long)Bsz * Hdim;        // B*V

  // 1) h_t = hidden @ W_h_w^T + W_h_b
  gemm_bias_kernel<<<dim3(Bsz / 32, Hdim / 64), 32, 0, stream>>>(
      hid, Whw, Whb, ht, Hdim, Hdim);

  // 2) fused attention scores (34.4 GFLOP WMMA; proj never materialized;
  //    deterministic partial sums, no FP atomics)
  attn_scores_kernel<<<dim3(HWd / 16, Hdim / 64, Bsz), 32, 0, stream>>>(
      img, ht, WFw, WFb, part);

  // 3) softmax over spatial positions (folds the 8-way partial reduce)
  attn_softmax_kernel<<<Bsz, HWd, 0, stream>>>(part, attn);

  // 4) context vector
  context_kernel<<<Bsz, 256, 0, stream>>>(img, attn, ctx);

  // 5) LSTM gates (fused concat: tok + ctx through W_ih split, hid through W_hh)
  gates_kernel<<<dim3(Bsz / 32, (4 * Hdim) / 64), 32, 0, stream>>>(
      tok, ctx, hid, Wih, bih, Whh, bhh, gates);

  // 6) LSTM cell elementwise -> h_new, c_new (directly into d_out)
  lstm_kernel<<<(Bsz * Hdim + 255) / 256, 256, 0, stream>>>(gates, cell, hNew, cNew);

  // 7) vocab head logits (8.4 GFLOP WMMA; W_out L2-resident across 8 M-tiles)
  gemm_bias_kernel<<<dim3(Bsz / 32, Vdim / 64), 32, 0, stream>>>(
      hNew, Wout, bout, logits, Vdim, Hdim);

  // 8) softmax over vocabulary
  vocab_softmax_kernel<<<Bsz, 1024, 0, stream>>>(logits, probs);
}
